// GCN_3728031613302
// MI455X (gfx1250) — matrix operand order repaired and checked by reading, not compiled, not run on hardware
//
#include <hip/hip_runtime.h>
#include <hip/hip_bf16.h>

// ---------------------------------------------------------------------------
// 2-layer GCN forward for gfx1250 (MI455X, wave32, WMMA).
//   h1  = relu( Dsym (X W1) + b1 )
//   out =       Dsym (h1 W2) + b2
// GEMMs: v_wmma_f32_16x16x32_bf16, operands pre-converted to bf16 (hardware
// cvt, done once), W pre-transposed so A and B fragments are contiguous
// 16-byte global_load_b128 chunks. Accumulation stays f32.
// Scatter: global_atomic_add_f32 with float4 gathers; ReLU fused into the
// layer-2 operand conversion.
// ---------------------------------------------------------------------------

#define N_NODES 50000
#define N_IN    128
#define N_HID   512
#define N_OUT   256
#define N_EDGES 500000

typedef __attribute__((ext_vector_type(16))) __bf16 v16bf;
typedef __attribute__((ext_vector_type(8)))  __bf16 v8bf;
typedef __attribute__((ext_vector_type(4)))  __bf16 v4bf;
typedef __attribute__((ext_vector_type(8)))  float  v8f;

// --------------------------- degree / norm ---------------------------------
__global__ void zero_f32_kernel(float* __restrict__ p, int n) {
    int i = blockIdx.x * blockDim.x + threadIdx.x;
    if (i < n) p[i] = 0.0f;
}

__global__ void degree_kernel(const int* __restrict__ tgt, float* __restrict__ deg, int E) {
    int i = blockIdx.x * blockDim.x + threadIdx.x;
    if (i < E) atomicAdd(&deg[tgt[i]], 1.0f);
}

// deg -> dinv in place: dinv = rsqrt(deg + 1)   (+1 = self loop)
__global__ void dinv_kernel(float* __restrict__ deg, int n) {
    int i = blockIdx.x * blockDim.x + threadIdx.x;
    if (i < n) deg[i] = rsqrtf(deg[i] + 1.0f);
}

// --------------------------- operand conversion ----------------------------
// f32 -> bf16 (hardware v_cvt), optional fused ReLU. n multiple of 4.
__global__ void cvt_bf16_kernel(const float* __restrict__ in, __bf16* __restrict__ out,
                                long long n, int do_relu) {
    long long i = ((long long)blockIdx.x * blockDim.x + threadIdx.x) * 4;
    if (i < n) {
        float4 v = *(const float4*)(in + i);
        if (do_relu) {
            v.x = fmaxf(v.x, 0.0f); v.y = fmaxf(v.y, 0.0f);
            v.z = fmaxf(v.z, 0.0f); v.w = fmaxf(v.w, 0.0f);
        }
        v4bf o;
        o[0] = (__bf16)v.x; o[1] = (__bf16)v.y;
        o[2] = (__bf16)v.z; o[3] = (__bf16)v.w;
        *(v4bf*)(out + i) = o;
    }
}

// W[K,N] (row-major f32) -> Wt[N,K] (row-major bf16). Tiny matrices, L2-resident.
__global__ void wtrans_bf16_kernel(const float* __restrict__ W, __bf16* __restrict__ Wt,
                                   int K, int N) {
    int idx = blockIdx.x * blockDim.x + threadIdx.x;
    if (idx < K * N) {
        int k = idx / N;
        int n = idx - k * N;                 // coalesced read over n
        Wt[(size_t)n * K + k] = (__bf16)W[idx];
    }
}

// --------------------------- WMMA GEMM -------------------------------------
// H[M,N] = A[M,K] * Wt[N,K]^T; A,Wt bf16; M = 50000 (3125x16), K%32==0, N%64==0.
// Block = 128 threads (4 waves); each wave owns one 16x16 tile (16x64/block).
// Per K-step: A = two contiguous 8xbf16 runs, B = one contiguous 16xbf16 run
// (W transposed), i.e. 4x global_load_b128 + 1x v_wmma per 8192 MACs.
__global__ __launch_bounds__(128) void gemm_bf16_wmma_kernel(
    const __bf16* __restrict__ A, const __bf16* __restrict__ Wt,
    float* __restrict__ H, int K, int N) {
    const int lane = threadIdx.x & 31;
    const int wave = threadIdx.x >> 5;
    const int half = lane >> 4;      // 0: lanes 0-15, 1: lanes 16-31
    const int lm   = lane & 15;
    const int m0   = blockIdx.x * 16;
    const int n0   = blockIdx.y * 64 + wave * 16;

    const __bf16* __restrict__ arow = A  + (size_t)(m0 + lm) * K;
    const __bf16* __restrict__ bcol = Wt + (size_t)(n0 + lm) * K;
    v8f c = {};

    for (int k0 = 0; k0 < K; k0 += 32) {
        __builtin_prefetch(arow + k0 + 32, 0, 1);   // global_prefetch_b8
        // A fragment (16-bit A 16x32): lane (half,lm) holds
        // K = 16*half..16*half+15 in elems 0..15 (same order as B).
        v8bf alo = *(const v8bf*)(arow + k0 + 16 * half);
        v8bf ahi = *(const v8bf*)(arow + k0 + 16 * half + 8);
        v16bf a;
        #pragma unroll
        for (int i = 0; i < 8; ++i) { a[i] = alo[i]; a[i + 8] = ahi[i]; }
        // B fragment (32x16): elem e holds K = 16*half + e at column lm;
        // contiguous in K thanks to the transpose.
        v16bf b = *(const v16bf*)(bcol + k0 + 16 * half);
        c = __builtin_amdgcn_wmma_f32_16x16x32_bf16(
                /*neg_a=*/false, a, /*neg_b=*/false, b,
                /*c_mod=*/(short)0, c, /*reuse_a=*/false, /*reuse_b=*/false);
    }
    // C/D layout: VGPR r -> row r + 8*half, col lm
    #pragma unroll
    for (int r = 0; r < 8; ++r) {
        int m = m0 + r + 8 * half;
        H[(size_t)m * N + (n0 + lm)] = c[r];
    }
}

// --------------------------- aggregation -----------------------------------
// agg[i] = dinv[i]^2 * h[i] + bias      (self-loop term + bias as init)
__global__ void init_agg_kernel(const float* __restrict__ h,
                                const float* __restrict__ dinv,
                                const float* __restrict__ bias,
                                float* __restrict__ agg, int F) {
    int node = blockIdx.x;
    int f4   = threadIdx.x * 4;
    float d  = dinv[node];
    d = d * d;
    const float4 hv = *(const float4*)(h + (size_t)node * F + f4);
    const float4 bv = *(const float4*)(bias + f4);
    float4 o;
    o.x = d * hv.x + bv.x;  o.y = d * hv.y + bv.y;
    o.z = d * hv.z + bv.z;  o.w = d * hv.w + bv.w;
    *(float4*)(agg + (size_t)node * F + f4) = o;
}

// one block per edge: agg[tgt] += dinv[src]*dinv[tgt] * h[src]
__global__ void scatter_kernel(const int* __restrict__ src,
                               const int* __restrict__ tgt,
                               const float* __restrict__ dinv,
                               const float* __restrict__ h,
                               float* __restrict__ agg, int F) {
    int e = blockIdx.x;
    int s = src[e];
    int t = tgt[e];
    float w = dinv[s] * dinv[t];
    const float* __restrict__ hs = h + (size_t)s * F;
    float* __restrict__ at = agg + (size_t)t * F;
    int f4 = threadIdx.x * 4;
    float4 v = *(const float4*)(hs + f4);
    atomicAdd(&at[f4 + 0], w * v.x);
    atomicAdd(&at[f4 + 1], w * v.y);
    atomicAdd(&at[f4 + 2], w * v.z);
    atomicAdd(&at[f4 + 3], w * v.w);
}

// --------------------------- launch ----------------------------------------
extern "C" void kernel_launch(void* const* d_in, const int* in_sizes, int n_in,
                              void* d_out, int out_size, void* d_ws, size_t ws_size,
                              hipStream_t stream) {
    const float* x    = (const float*)d_in[0];
    const int*   edge = (const int*)d_in[1];     // [2, E]: row0=src, row1=tgt
    const float* W1   = (const float*)d_in[2];
    const float* b1   = (const float*)d_in[3];
    const float* W2   = (const float*)d_in[4];
    const float* b2   = (const float*)d_in[5];
    float*       out  = (float*)d_out;

    const int* src = edge;
    const int* tgt = edge + N_EDGES;

    // workspace layout:
    //   dinv f32[50000] | h f32[50000*512] | agg1 f32[50000*512]
    // | Abf bf16[50000*512] (shared: X_bf then relu(agg1)_bf) | W1t | W2t
    float*  dinv = (float*)d_ws;
    float*  h    = dinv + N_NODES;
    float*  agg1 = h + (size_t)N_NODES * N_HID;
    __bf16* Abf  = (__bf16*)(agg1 + (size_t)N_NODES * N_HID);
    __bf16* W1t  = Abf + (size_t)N_NODES * N_HID;
    __bf16* W2t  = W1t + (size_t)N_IN * N_HID;

    // ---- normalization ----
    zero_f32_kernel<<<(N_NODES + 255) / 256, 256, 0, stream>>>(dinv, N_NODES);
    degree_kernel<<<(N_EDGES + 255) / 256, 256, 0, stream>>>(tgt, dinv, N_EDGES);
    dinv_kernel<<<(N_NODES + 255) / 256, 256, 0, stream>>>(dinv, N_NODES);

    // ---- operand prep for layer 1 ----
    {
        long long n = (long long)N_NODES * N_IN;
        cvt_bf16_kernel<<<(unsigned)((n / 4 + 255) / 256), 256, 0, stream>>>(
            x, Abf, n, /*relu=*/0);
        wtrans_bf16_kernel<<<(N_IN * N_HID + 255) / 256, 256, 0, stream>>>(
            W1, W1t, N_IN, N_HID);
    }

    // ---- layer 1: h = X @ W1 ; agg1 = Dsym-scatter(h) + b1 ----
    gemm_bf16_wmma_kernel<<<dim3(N_NODES / 16, N_HID / 64), 128, 0, stream>>>(
        Abf, W1t, h, N_IN, N_HID);
    init_agg_kernel<<<N_NODES, N_HID / 4, 0, stream>>>(h, dinv, b1, agg1, N_HID);
    scatter_kernel<<<N_EDGES, N_HID / 4, 0, stream>>>(src, tgt, dinv, h, agg1, N_HID);

    // ---- operand prep for layer 2 (ReLU fused into conversion) ----
    {
        long long n = (long long)N_NODES * N_HID;
        cvt_bf16_kernel<<<(unsigned)((n / 4 + 255) / 256), 256, 0, stream>>>(
            agg1, Abf, n, /*relu=*/1);
        wtrans_bf16_kernel<<<(N_HID * N_OUT + 255) / 256, 256, 0, stream>>>(
            W2, W2t, N_HID, N_OUT);
    }

    // ---- layer 2: h2 = relu(agg1) @ W2 (reuse h) ; out = Dsym-scatter(h2) + b2 ----
    gemm_bf16_wmma_kernel<<<dim3(N_NODES / 16, N_OUT / 64), 128, 0, stream>>>(
        Abf, W2t, h, N_HID, N_OUT);
    init_agg_kernel<<<N_NODES, N_OUT / 4, 0, stream>>>(h, dinv, b2, out, N_OUT);
    scatter_kernel<<<N_EDGES, N_OUT / 4, 0, stream>>>(src, tgt, dinv, h, out, N_OUT);
}
